// RouterAKT_1735166788142
// MI455X (gfx1250) — matrix operand order repaired and checked by reading, not compiled, not run on hardware
//
#include <hip/hip_runtime.h>
#include <hip/hip_bf16.h>
#include <math.h>

// ---------------------------------------------------------------------------
// RouterAKT forward for MI455X (gfx1250).
// Compute-bound (~135 GFLOP GEMMs vs tens of MB of traffic; all L2-resident
// at 192MB, HBM 23.3TB/s never limits). Every matmul uses
// v_wmma_f32_16x16x32_bf16 (bf16 A/B, f32 accum). Softmax/LN/residual/router
// stay in fp32. Attention is flash-style online softmax with 64-key tiles
// (one cross-lane reduction pass amortized over 8 WMMAs); P tiles go through
// LDS to convert C-layout -> A-fragment layout.
// ---------------------------------------------------------------------------

typedef __bf16 bf16;
typedef bf16  v16bf __attribute__((ext_vector_type(16)));
typedef bf16  v8bf  __attribute__((ext_vector_type(8)));
typedef float v8f   __attribute__((ext_vector_type(8)));

#define BB   16
#define SS   512
#define EE   256
#define HH   8
#define DK_  32
#define MTOK (BB * SS)          // 8192 tokens
#define LDEP 12
#define NDYN 6
#define NSEL 4
#define FC1_ 512
#define FC2_ 256

union AFrag { v16bf v; v8bf h[2]; };

// ---------------------------------------------------------------------------
// Generic bf16 WMMA GEMM: C[M,N] = A[M,K] (bf16, row-major) * Bt[N,K] (bf16,
// pre-transposed) + bias, optional ReLU, optional f32 and/or bf16 outputs.
// Block = 128 threads (4 waves, 2x2 arrangement), block tile 64x64,
// wave tile 32x32 via 2x2 v_wmma_f32_16x16x32_bf16 accumulators.
// ---------------------------------------------------------------------------
__global__ __launch_bounds__(128)
void gemm_bf16_wmma(const bf16* __restrict__ A, const bf16* __restrict__ Bt,
                    const float* __restrict__ bias, float* __restrict__ Cf,
                    bf16* __restrict__ Cb, int M, int N, int K, int relu)
{
    const int lane = threadIdx.x & 31;
    const int wave = threadIdx.x >> 5;
    const int ln   = lane & 15;
    const int half = lane >> 4;
    const int n0 = blockIdx.x * 64 + (wave & 1) * 32;
    const int m0 = blockIdx.y * 64 + (wave >> 1) * 32;

    // CDNA5 prefetch hint for the streamed rows (emits global_prefetch_b8).
    __builtin_prefetch(A + (size_t)(m0 + ln) * K, 0, 3);
    __builtin_prefetch(Bt + (size_t)(n0 + ln) * K, 0, 3);

    v8f acc[2][2];
    for (int i = 0; i < 2; i++)
        for (int j = 0; j < 2; j++)
            for (int q = 0; q < 8; q++) acc[i][j][q] = 0.f;

    for (int kk = 0; kk < K; kk += 32) {
        AFrag a[2], b[2];
        // A fragment: lanes 0-15 hold K [kk+0..15] of row m,
        // lanes 16-31 hold K [kk+16..31] (contiguous, matches B).
#pragma unroll
        for (int i = 0; i < 2; i++) {
            const bf16* ap = A + (size_t)(m0 + i * 16 + ln) * K + kk;
            a[i].h[0] = *(const v8bf*)(ap + 16 * half);
            a[i].h[1] = *(const v8bf*)(ap + 16 * half + 8);
        }
        // B fragment: lane = column, lanes 0-15 K [kk..kk+15], 16-31 K [+16..31].
#pragma unroll
        for (int j = 0; j < 2; j++) {
            const bf16* bp = Bt + (size_t)(n0 + j * 16 + ln) * K + kk + 16 * half;
            b[j].v = *(const v16bf*)bp;
        }
#pragma unroll
        for (int i = 0; i < 2; i++)
#pragma unroll
            for (int j = 0; j < 2; j++)
                acc[i][j] = __builtin_amdgcn_wmma_f32_16x16x32_bf16(
                    false, a[i].v, false, b[j].v, (short)0, acc[i][j], false, false);
    }

#pragma unroll
    for (int i = 0; i < 2; i++)
        for (int j = 0; j < 2; j++)
            for (int v = 0; v < 8; v++) {
                const int row = m0 + i * 16 + v + 8 * half;
                const int col = n0 + j * 16 + ln;
                float x = acc[i][j][v];
                if (bias) x += bias[col];
                if (relu) x = fmaxf(x, 0.f);
                const size_t o = (size_t)row * N + col;
                if (Cf) Cf[o] = x;
                if (Cb) Cb[o] = (bf16)x;
            }
}

// ---------------------------------------------------------------------------
// Flash-style attention, K==Q (kq_same + shared Wq). One wave = 16 query rows
// of one (b,h). Key tiles of 64: 4 score WMMAs, one online-softmax reduction
// pass (half-wave shfls; row stats live per half), P (16x64) staged to LDS as
// bf16 and re-loaded as two K=32 A-fragments, P@V via 4 WMMAs against
// transposed V. Gating + 1/l normalization fused into the bf16 ctx write.
// ---------------------------------------------------------------------------
__global__ __launch_bounds__(128)
void attn_wmma(const bf16* __restrict__ qh, const bf16* __restrict__ vt,
               const float* __restrict__ hg, bf16* __restrict__ ctx, int strict)
{
    __shared__ __align__(16) bf16 plds[4][16][64];
    const int lane = threadIdx.x & 31;
    const int wave = threadIdx.x >> 5;
    const int gw = blockIdx.x * 4 + wave;
    const int qt = gw & 31;
    const int h  = (gw >> 5) & 7;
    const int b  = gw >> 8;
    const int q0 = qt * 16;
    const int ln = lane & 15;
    const int half = lane >> 4;
    const float RS = 0.17677669529663689f;   // 1/sqrt(DK)

    AFrag aq;
    {
        const bf16* qp = qh + ((size_t)(b * SS + q0 + ln) * EE + h * 32);
        aq.h[0] = *(const v8bf*)(qp + 16 * half);
        aq.h[1] = *(const v8bf*)(qp + 16 * half + 8);
    }

    float mrow[8], lrow[8];
    v8f o0, o1;
    for (int v = 0; v < 8; v++) { mrow[v] = -3.0e38f; lrow[v] = 0.f; o0[v] = 0.f; o1[v] = 0.f; }

    const int jmax = ((q0 + 15) >> 6) << 6;        // last 64-key block touching diag
    for (int j0 = 0; j0 <= jmax; j0 += 64) {
        // ---- scores: 4x v_wmma over 64 keys ----
        v8f s[4];
#pragma unroll
        for (int c = 0; c < 4; c++) {
            AFrag bk;
            bk.v = *(const v16bf*)(qh + ((size_t)(b * SS + j0 + c * 16 + ln) * EE + h * 32 + 16 * half));
            for (int v = 0; v < 8; v++) s[c][v] = 0.f;
            s[c] = __builtin_amdgcn_wmma_f32_16x16x32_bf16(false, aq.v, false, bk.v, (short)0, s[c], false, false);
        }

        // ---- online softmax: one reduction pass per 64 keys ----
#pragma unroll
        for (int v = 0; v < 8; v++) {
            const int irow = q0 + v + 8 * half;
            float xc[4];
#pragma unroll
            for (int c = 0; c < 4; c++) {
                const int jc = j0 + c * 16 + ln;
                const bool ok = strict ? (jc < irow) : (jc <= irow);
                xc[c] = ok ? s[c][v] * RS : -1.0e32f;
            }
            float t = fmaxf(fmaxf(xc[0], xc[1]), fmaxf(xc[2], xc[3]));
            t = fmaxf(t, __shfl_xor(t, 1, 32));
            t = fmaxf(t, __shfl_xor(t, 2, 32));
            t = fmaxf(t, __shfl_xor(t, 4, 32));
            t = fmaxf(t, __shfl_xor(t, 8, 32));
            const float mn = fmaxf(mrow[v], t);
            const float corr = __expf(mrow[v] - mn);
            float p[4], rs = 0.f;
#pragma unroll
            for (int c = 0; c < 4; c++) { p[c] = __expf(xc[c] - mn); rs += p[c]; }
            rs += __shfl_xor(rs, 1, 32);
            rs += __shfl_xor(rs, 2, 32);
            rs += __shfl_xor(rs, 4, 32);
            rs += __shfl_xor(rs, 8, 32);
            lrow[v] = lrow[v] * corr + rs;
            mrow[v] = mn;
            o0[v] *= corr; o1[v] *= corr;
#pragma unroll
            for (int c = 0; c < 4; c++)
                plds[wave][v + 8 * half][c * 16 + ln] = (bf16)p[c];
        }
        asm volatile("s_wait_dscnt 0" ::: "memory");   // LDS RAW: P staged -> A-frags

        // ---- P (16x64) as two K=32 A-fragments ----
        AFrag ap0, ap1;
        ap0.h[0] = *(const v8bf*)(&plds[wave][ln][16 * half]);
        ap0.h[1] = *(const v8bf*)(&plds[wave][ln][8 + 16 * half]);
        ap1.h[0] = *(const v8bf*)(&plds[wave][ln][32 + 16 * half]);
        ap1.h[1] = *(const v8bf*)(&plds[wave][ln][40 + 16 * half]);

        // ---- V B-fragments (transposed V: contiguous S) ----
        const bf16* vb = vt + ((size_t)(b * HH + h) * DK_) * SS;
        AFrag bv00, bv01, bv10, bv11;
        bv00.v = *(const v16bf*)(vb + (size_t)ln * SS + j0 + 16 * half);
        bv01.v = *(const v16bf*)(vb + (size_t)(16 + ln) * SS + j0 + 16 * half);
        bv10.v = *(const v16bf*)(vb + (size_t)ln * SS + j0 + 32 + 16 * half);
        bv11.v = *(const v16bf*)(vb + (size_t)(16 + ln) * SS + j0 + 32 + 16 * half);

        o0 = __builtin_amdgcn_wmma_f32_16x16x32_bf16(false, ap0.v, false, bv00.v, (short)0, o0, false, false);
        o1 = __builtin_amdgcn_wmma_f32_16x16x32_bf16(false, ap0.v, false, bv01.v, (short)0, o1, false, false);
        o0 = __builtin_amdgcn_wmma_f32_16x16x32_bf16(false, ap1.v, false, bv10.v, (short)0, o0, false, false);
        o1 = __builtin_amdgcn_wmma_f32_16x16x32_bf16(false, ap1.v, false, bv11.v, (short)0, o1, false, false);
    }

#pragma unroll
    for (int v = 0; v < 8; v++) {
        const int s = q0 + v + 8 * half;
        const size_t tok = (size_t)b * SS + s;
        const float g = hg[tok * 8 + h] / lrow[v];
        bf16* cp = ctx + tok * EE + h * 32;
        cp[ln]      = (bf16)(o0[v] * g);
        cp[16 + ln] = (bf16)(o1[v] * g);
    }
}

// --- weight prep: fp32 [cnt,K,N] -> bf16 [cnt,N,K] (transposed) -------------
__global__ void transconv_kernel(const float* __restrict__ in, bf16* __restrict__ out,
                                 int K, int N, long total)
{
    long idx = (long)blockIdx.x * blockDim.x + threadIdx.x;
    if (idx >= total) return;
    const long mn = (long)K * N;
    const long mat = idx / mn;
    const long r = idx - mat * mn;
    const int k = (int)(r / N), n = (int)(r - (long)k * N);
    out[mat * mn + (size_t)n * K + k] = (bf16)in[idx];
}

// --- embeddings: q_emb / qa_emb (+pid*diff), write f32 + bf16, seed x,y -----
__global__ void embed_kernel(const int* __restrict__ skills, const int* __restrict__ resp,
                             const int* __restrict__ quest,
                             const float* __restrict__ q_embed, const float* __restrict__ qa_embed,
                             const float* __restrict__ q_diff, const float* __restrict__ qa_diff,
                             const float* __restrict__ dpar,
                             float* __restrict__ qe_f, bf16* __restrict__ qe_b,
                             float* __restrict__ y_f, bf16* __restrict__ y_b,
                             float* __restrict__ x_f, bf16* __restrict__ x_b)
{
    const long idx = (long)blockIdx.x * blockDim.x + threadIdx.x;
    if (idx >= (long)MTOK * EE) return;
    const long t = idx / EE;
    const int e = (int)(idx - t * EE);
    const int sk = skills[t];
    int r = resp[t]; r = (r > -1) ? r : 0;
    const float pid = dpar[quest[t]];
    float q  = q_embed[(size_t)sk * EE + e];
    float qa = q + qa_embed[(size_t)r * EE + e];
    const float qd  = q_diff[(size_t)sk * EE + e];
    const float qad = qa_diff[(size_t)(sk + 1000 * r) * EE + e];
    q  += pid * qd;
    qa += pid * (qad + qd);
    qe_f[idx] = q;  qe_b[idx] = (bf16)q;
    x_f[idx]  = q;  x_b[idx]  = (bf16)q;
    y_f[idx]  = qa; y_b[idx]  = (bf16)qa;
}

// --- MoH router: logits = x @ Wg [E,6], hard top-4 one-hot + 2 shared heads -
__global__ void router_kernel(const float* __restrict__ X, const float* __restrict__ Wg,
                              float* __restrict__ hg, int M)
{
    const int t = blockIdx.x * blockDim.x + threadIdx.x;
    if (t >= M) return;
    const float* x = X + (size_t)t * EE;
    float lg[NDYN];
    for (int d = 0; d < NDYN; d++) lg[d] = 0.f;
    for (int e = 0; e < EE; e++) {
        const float xv = x[e];
        for (int d = 0; d < NDYN; d++) lg[d] += xv * Wg[(size_t)e * NDYN + d];
    }
    float* o = hg + (size_t)t * 8;
    o[0] = 1.f; o[1] = 1.f;
    for (int d = 0; d < NDYN; d++) {
        int rank = 0;
        for (int d2 = 0; d2 < NDYN; d2++)
            rank += (lg[d2] > lg[d]) || (lg[d2] == lg[d] && d2 < d);
        o[2 + d] = (rank < NSEL) ? 1.f : 0.f;
    }
}

// --- V [B,S,H,DK] bf16 -> Vt [B,H,DK,S] bf16 (contiguous S for B-fragments) -
__global__ void transposeV_kernel(const bf16* __restrict__ v, bf16* __restrict__ vt)
{
    const long idx = (long)blockIdx.x * blockDim.x + threadIdx.x;
    if (idx >= (long)MTOK * EE) return;
    const long t = idx / EE;
    const int c = (int)(idx - t * EE);
    const int h = c >> 5, d = c & 31;
    const int b = (int)(t / SS), s = (int)(t - (long)b * SS);
    vt[(((size_t)(b * HH + h) * DK_) + d) * SS + s] = v[idx];
}

// --- residual add + LayerNorm, dual f32/bf16 output -------------------------
__global__ __launch_bounds__(256)
void add_ln_kernel(const float* __restrict__ A, const float* __restrict__ Bv,
                   const float* __restrict__ g, const float* __restrict__ be,
                   float* __restrict__ of, bf16* __restrict__ ob)
{
    __shared__ float sm[256];
    const int t = blockIdx.x, e = threadIdx.x;
    const size_t o = (size_t)t * EE + e;
    const float x = A[o] + Bv[o];
    sm[e] = x; __syncthreads();
    for (int s = 128; s > 0; s >>= 1) { if (e < s) sm[e] += sm[e + s]; __syncthreads(); }
    const float mean = sm[0] / EE; __syncthreads();
    const float dx = x - mean;
    sm[e] = dx * dx; __syncthreads();
    for (int s = 128; s > 0; s >>= 1) { if (e < s) sm[e] += sm[e + s]; __syncthreads(); }
    const float var = sm[0] / EE;
    const float y = dx * rsqrtf(var + 1e-5f) * g[e] + be[e];
    of[o] = y; ob[o] = (bf16)y;
}

// --- h = concat(x, q_emb) as bf16 [M, 2E] -----------------------------------
__global__ void assemble_h_kernel(const bf16* __restrict__ x, const bf16* __restrict__ qe,
                                  bf16* __restrict__ h)
{
    const long idx = (long)blockIdx.x * blockDim.x + threadIdx.x;
    if (idx >= (long)MTOK * 2 * EE) return;
    const long t = idx / (2 * EE);
    const int c = (int)(idx - t * 2 * EE);
    h[idx] = (c < EE) ? x[t * EE + c] : qe[t * EE + (c - EE)];
}

// --- final: sigmoid(h2 @ W3 + b3), drop s==0 --------------------------------
__global__ void final_kernel(const float* __restrict__ h2, const float* __restrict__ W3,
                             const float* __restrict__ b3, float* __restrict__ out, int M)
{
    const int t = blockIdx.x * blockDim.x + threadIdx.x;
    if (t >= M) return;
    const float* hh = h2 + (size_t)t * FC2_;
    float acc = b3[0];
    for (int e = 0; e < FC2_; e++) acc += hh[e] * W3[e];
    const float y = 1.f / (1.f + __expf(-acc));
    const int b = t / SS, s = t - b * SS;
    if (s > 0) out[(size_t)b * (SS - 1) + (s - 1)] = y;
}

// ---------------------------------------------------------------------------
extern "C" void kernel_launch(void* const* d_in, const int* in_sizes, int n_in,
                              void* d_out, int out_size, void* d_ws, size_t ws_size,
                              hipStream_t stream)
{
    const int* skills    = (const int*)d_in[0];
    const int* responses = (const int*)d_in[1];
    const int* questions = (const int*)d_in[2];
    const float* q_embed  = (const float*)d_in[4];
    const float* qa_embed = (const float*)d_in[5];
    const float* q_diff   = (const float*)d_in[6];
    const float* qa_diff  = (const float*)d_in[7];
    const float* dpar     = (const float*)d_in[8];
    const float* Wq = (const float*)d_in[9];   const float* bq = (const float*)d_in[10];
    const float* Wv = (const float*)d_in[11];  const float* bv = (const float*)d_in[12];
    const float* Wo = (const float*)d_in[13];  const float* bo = (const float*)d_in[14];
    const float* Wg = (const float*)d_in[15];
    const float* l1g = (const float*)d_in[16]; const float* l1b = (const float*)d_in[17];
    const float* l2g = (const float*)d_in[18]; const float* l2b = (const float*)d_in[19];
    const float* fW1 = (const float*)d_in[20]; const float* fb1 = (const float*)d_in[21];
    const float* fW2 = (const float*)d_in[22]; const float* fb2 = (const float*)d_in[23];
    const float* oW1 = (const float*)d_in[24]; const float* ob1 = (const float*)d_in[25];
    const float* oW2 = (const float*)d_in[26]; const float* ob2 = (const float*)d_in[27];
    const float* oW3 = (const float*)d_in[28]; const float* ob3 = (const float*)d_in[29];
    float* out = (float*)d_out;

    // bump allocator over workspace
    uint8_t* pw = (uint8_t*)d_ws;
    auto alloc = [&](size_t n) { uint8_t* r = pw; pw += (n + 255) & ~(size_t)255; return (void*)r; };
    const size_t WSZ = (size_t)LDEP * EE * EE;    // per weight stack (elements)
    bf16* wqT  = (bf16*)alloc(WSZ * 2);
    bf16* wvT  = (bf16*)alloc(WSZ * 2);
    bf16* woT  = (bf16*)alloc(WSZ * 2);
    bf16* f1T  = (bf16*)alloc(WSZ * 2);
    bf16* f2T  = (bf16*)alloc(WSZ * 2);
    bf16* ow1T = (bf16*)alloc((size_t)2 * EE * FC1_ * 2);
    bf16* ow2T = (bf16*)alloc((size_t)FC1_ * FC2_ * 2);
    const size_t AF = (size_t)MTOK * EE;
    float* qe_f = (float*)alloc(AF * 4);  bf16* qe_b = (bf16*)alloc(AF * 2);
    float* y_f  = (float*)alloc(AF * 4);  bf16* y_b  = (bf16*)alloc(AF * 2);
    float* x_f  = (float*)alloc(AF * 4);  bf16* x_b  = (bf16*)alloc(AF * 2);
    bf16* qh_b  = (bf16*)alloc(AF * 2);
    bf16* vh_b  = (bf16*)alloc(AF * 2);
    bf16* vt_b  = (bf16*)alloc(AF * 2);
    bf16* ctx_b = (bf16*)alloc(AF * 2);
    float* ao_f = (float*)alloc(AF * 4);
    float* t_f  = (float*)alloc(AF * 4);  bf16* t_b = (bf16*)alloc(AF * 2);
    bf16* ffn1_b = (bf16*)alloc(AF * 2);
    float* ffn2_f = (float*)alloc(AF * 4);
    float* hg_f = (float*)alloc((size_t)MTOK * 8 * 4);
    bf16* h_b  = (bf16*)alloc((size_t)MTOK * 2 * EE * 2);
    bf16* h1_b = (bf16*)alloc((size_t)MTOK * FC1_ * 2);
    float* h2_f = (float*)alloc((size_t)MTOK * FC2_ * 4);

    // ---- weight prep (transpose + bf16) ----
    {
        const long tot = (long)LDEP * EE * EE;
        const int blk = (int)((tot + 255) / 256);
        transconv_kernel<<<blk, 256, 0, stream>>>(Wq,  wqT, EE, EE, tot);
        transconv_kernel<<<blk, 256, 0, stream>>>(Wv,  wvT, EE, EE, tot);
        transconv_kernel<<<blk, 256, 0, stream>>>(Wo,  woT, EE, EE, tot);
        transconv_kernel<<<blk, 256, 0, stream>>>(fW1, f1T, EE, EE, tot);
        transconv_kernel<<<blk, 256, 0, stream>>>(fW2, f2T, EE, EE, tot);
        const long t1 = (long)2 * EE * FC1_;
        transconv_kernel<<<(int)((t1 + 255) / 256), 256, 0, stream>>>(oW1, ow1T, 2 * EE, FC1_, t1);
        const long t2 = (long)FC1_ * FC2_;
        transconv_kernel<<<(int)((t2 + 255) / 256), 256, 0, stream>>>(oW2, ow2T, FC1_, FC2_, t2);
    }

    // ---- embeddings ----
    embed_kernel<<<(int)((AF + 255) / 256), 256, 0, stream>>>(
        skills, responses, questions, q_embed, qa_embed, q_diff, qa_diff, dpar,
        qe_f, qe_b, y_f, y_b, x_f, x_b);

    auto gemm = [&](const bf16* A, const bf16* Bt, const float* bias,
                    float* Cf, bf16* Cb, int M_, int N_, int K_, int relu) {
        dim3 g(N_ / 64, M_ / 64);
        gemm_bf16_wmma<<<g, 128, 0, stream>>>(A, Bt, bias, Cf, Cb, M_, N_, K_, relu);
    };

    auto layer = [&](int i, float* inf, bf16* inb, const bf16* vinb, int strict) {
        const size_t wo = (size_t)i * EE * EE;
        gemm(inb,  wqT + wo, bq + (size_t)i * EE, nullptr, qh_b, MTOK, EE, EE, 0);   // Q (==K)
        gemm(vinb, wvT + wo, bv + (size_t)i * EE, nullptr, vh_b, MTOK, EE, EE, 0);   // V
        transposeV_kernel<<<(int)(AF / 256), 256, 0, stream>>>(vh_b, vt_b);
        router_kernel<<<MTOK / 256, 256, 0, stream>>>(inf, Wg + (size_t)i * EE * NDYN, hg_f, MTOK);
        attn_wmma<<<(BB * HH * (SS / 16)) / 4, 128, 0, stream>>>(qh_b, vt_b, hg_f, ctx_b, strict);
        gemm(ctx_b, woT + wo, bo + (size_t)i * EE, ao_f, nullptr, MTOK, EE, EE, 0);  // Wo
        add_ln_kernel<<<MTOK, 256, 0, stream>>>(inf, ao_f, l1g + (size_t)i * EE, l1b + (size_t)i * EE, t_f, t_b);
        gemm(t_b,    f1T + wo, fb1 + (size_t)i * EE, nullptr, ffn1_b, MTOK, EE, EE, 1); // FFN1+ReLU
        gemm(ffn1_b, f2T + wo, fb2 + (size_t)i * EE, ffn2_f, nullptr, MTOK, EE, EE, 0); // FFN2
        add_ln_kernel<<<MTOK, 256, 0, stream>>>(t_f, ffn2_f, l2g + (size_t)i * EE, l2b + (size_t)i * EE, inf, inb);
    };

    // blocks_1: y self-attention (causal)
    for (int i = 0; i < 4; i++) layer(i, y_f, y_b, y_b, 0);
    // blocks_2: x self (causal) then x/x/y cross (strict)
    int li = 4;
    for (int k = 0; k < 4; k++) {
        layer(li, x_f, x_b, x_b, 0); li++;
        layer(li, x_f, x_b, y_b, 1); li++;
    }

    // output MLP
    assemble_h_kernel<<<(int)(((size_t)MTOK * 2 * EE + 255) / 256), 256, 0, stream>>>(x_b, qe_b, h_b);
    gemm(h_b,  ow1T, ob1, nullptr, h1_b, MTOK, FC1_, 2 * EE, 1);
    gemm(h1_b, ow2T, ob2, h2_f, nullptr, MTOK, FC2_, FC1_, 1);
    final_kernel<<<MTOK / 256, 256, 0, stream>>>(h2_f, oW3, ob3, out, MTOK);
}
